// PairVelKernel_18056042512836
// MI455X (gfx1250) — matrix operand order repaired and checked by reading, not compiled, not run on hardware
//
#include <hip/hip_runtime.h>
#include <stdint.h>

typedef __attribute__((ext_vector_type(16))) _Float16 v16h;
typedef __attribute__((ext_vector_type(8)))  float    v8f;

#define N_EDGES_C 3200000
#define N_NODES_C 100000
#define HIDDEN_C  64
#define FEAT_IN_C 13
#define TILES_C   (N_EDGES_C / 16)

// Compiler-level memory fence: LDS data moves BETWEEN lanes of a wave, which
// per-thread alias analysis cannot see. This blocks CSE/DSE of ds ops across
// phase boundaries; emits no instructions (HW keeps LDS in-order per wave).
#define LDS_PHASE_FENCE() asm volatile("" ::: "memory")

union HFrag { v16h v; _Float16 h[16]; uint32_t u[8]; };
union FAcc  { v8f  v; float f[8]; };
union FeatPack { _Float16 h[32]; uint32_t u[16]; };

__global__ void zero_out_kernel(float* __restrict__ out, int n) {
    int i = blockIdx.x * blockDim.x + threadIdx.x;
    if (i < n) out[i] = 0.0f;
}

__launch_bounds__(256, 1)
__global__ void pairvel_wmma_kernel(
    const float* __restrict__ rel,   const int* __restrict__ tgt,
    const int*  __restrict__ src,    const float* __restrict__ force,
    const float* __restrict__ visc,  const float* __restrict__ med_,
    const float* __restrict__ cd_,   const float* __restrict__ W1,
    const float* __restrict__ b1,    const float* __restrict__ W2,
    const float* __restrict__ b2,    const float* __restrict__ W3,
    const float* __restrict__ b3,    float* __restrict__ out)
{
    // Per-wave private LDS slice: 16 rows x 64 f16 columns (2 KB). 8 waves/block.
    __shared__ __align__(16) _Float16 smem[8][16 * 64];

    const int lane      = threadIdx.x & 31;
    const int waveInBlk = threadIdx.x >> 5;
    const int nWaves    = gridDim.x * (blockDim.x >> 5);
    const int waveGlob  = blockIdx.x * (blockDim.x >> 5) + waveInBlk;

    _Float16* myLds = &smem[waveInBlk][0];

    const int  nlo    = lane & 15;
    const bool hiHalf = lane >= 16;

    // ---------------- scalars / biases (uniform-ish, L2-resident) ----------------
    const float median  = med_[0];
    const float cdist   = cd_[0];
    const float invVisc = 1.0f / visc[0];
    float b1v[4], b2v[4];
    #pragma unroll
    for (int j = 0; j < 4; ++j) { b1v[j] = b1[16 * j + nlo]; b2v[j] = b2[16 * j + nlo]; }
    const float b3v = (nlo < 3) ? b3[nlo] : 0.0f;

    // ---------------- weight B-fragments (contiguous k = 16h + s, 32x16 tile) -------
    // lane L: column n = nOff + (L%16); K base = kOff + (L>=16 ? 16 : 0);
    // VGPR v holds K = base+2v (lo half), base+2v+1 (hi half).
    auto loadB = [&](const float* W, int Ndim, int kOff, int nOff,
                     int kLim, int nLim) -> HFrag {
        HFrag f;
        const int n  = nOff + nlo;
        const int kb = kOff + (hiHalf ? 16 : 0);
        #pragma unroll
        for (int v = 0; v < 8; ++v) {
            const int k0 = kb + 2 * v;
            const int k1 = k0 + 1;
            float w0 = (k0 < kLim && n < nLim) ? W[k0 * Ndim + n] : 0.0f;
            float w1 = (k1 < kLim && n < nLim) ? W[k1 * Ndim + n] : 0.0f;
            f.h[2 * v]     = (_Float16)w0;
            f.h[2 * v + 1] = (_Float16)w1;
        }
        return f;
    };

    HFrag w1f[4], w2f[2][4], w3f[2];
    #pragma unroll
    for (int j = 0; j < 4; ++j) w1f[j] = loadB(W1, 64, 0, 16 * j, FEAT_IN_C, 64);
    #pragma unroll
    for (int t = 0; t < 2; ++t)
        #pragma unroll
        for (int j = 0; j < 4; ++j) w2f[t][j] = loadB(W2, 64, 32 * t, 16 * j, 64, 64);
    #pragma unroll
    for (int t = 0; t < 2; ++t) w3f[t] = loadB(W3, 3, 32 * t, 0, 64, 3);

    // ---------------- A-fragment reader (contiguous k = 16h + s, 16x32 tile) --------
    // lane L: row M = L%16; base = (L>=16 ? 16 : 0);
    // VGPR v: K pair = base + 2v  (pairs are K-consecutive -> b32 read)
    auto loadA = [&](int kOff) -> HFrag {
        HFrag f;
        const int base = hiHalf ? 16 : 0;
        const uint32_t* p = (const uint32_t*)(myLds + nlo * 64);
        #pragma unroll
        for (int v = 0; v < 8; ++v) {
            const int k0 = base + 2 * v;
            f.u[v] = p[(kOff + k0) >> 1];
        }
        return f;
    };

    // ---------------- main grid-stride loop over 16-edge tiles -------------------
    for (int tile = waveGlob; tile < TILES_C; tile += nWaves) {
        const int e0 = tile * 16;

        // Loop-carried WAR: this tile's feature stores vs. previous tile's A loads.
        LDS_PHASE_FENCE();

        // --- stage edge features (f16, K=0..12 valid, 13..31 zero) into LDS ---
        if (!hiHalf) {
            const int e = e0 + nlo;
            float rx = 0.f, ry = 0.f, rz = 0.f;
            float ft0 = 0.f, ft1 = 0.f, ft2 = 0.f, fs0 = 0.f, fs1 = 0.f, fs2 = 0.f;
            if (e < N_EDGES_C) {
                rx = rel[3 * e]; ry = rel[3 * e + 1]; rz = rel[3 * e + 2];
                const int ti = tgt[e], si = src[e];
                ft0 = force[3 * ti]; ft1 = force[3 * ti + 1]; ft2 = force[3 * ti + 2];
                fs0 = force[3 * si]; fs1 = force[3 * si + 1]; fs2 = force[3 * si + 2];
                // prefetch next tile's edge data (global_prefetch_b8)
                const int en = e + nWaves * 16;
                if (en < N_EDGES_C) {
                    __builtin_prefetch(&rel[3 * en], 0, 1);
                    __builtin_prefetch(&tgt[en], 0, 1);
                    __builtin_prefetch(&src[en], 0, 1);
                }
            }
            float dist = sqrtf(rx * rx + ry * ry + rz * rz);
            dist = fmaxf(dist, 1e-8f);
            const float rs    = dist - median;
            const float rsq   = rs * rs;
            const float rquad = rsq * rsq;
            const float mind  = dist - cdist;

            FeatPack fp;
            fp.h[0] = (_Float16)rx;   fp.h[1] = (_Float16)ry;  fp.h[2]  = (_Float16)rz;
            fp.h[3] = (_Float16)dist; fp.h[4] = (_Float16)rsq; fp.h[5]  = (_Float16)rquad;
            fp.h[6] = (_Float16)mind;
            fp.h[7] = (_Float16)ft0;  fp.h[8] = (_Float16)ft1; fp.h[9]  = (_Float16)ft2;
            fp.h[10] = (_Float16)fs0; fp.h[11] = (_Float16)fs1; fp.h[12] = (_Float16)fs2;
            #pragma unroll
            for (int k = 13; k < 32; ++k) fp.h[k] = (_Float16)0.0f;

            uint32_t* q = (uint32_t*)(myLds + nlo * 64);
            #pragma unroll
            for (int c = 0; c < 16; ++c) q[c] = fp.u[c];
        }

        // Write phase (features) -> read phase (xa).
        LDS_PHASE_FENCE();

        // --- layer 1: h1 = relu(x @ W1 + b1); x is 16x32 (K padded) -------------
        const HFrag xa = loadA(0);
        #pragma unroll
        for (int j = 0; j < 4; ++j) {
            v8f c = {};
            c = __builtin_amdgcn_wmma_f32_16x16x32_f16(false, xa.v, false, w1f[j].v,
                                                       (short)0, c, false, false);
            FAcc a; a.v = c;
            const int col   = 16 * j + nlo;
            const int mbase = hiHalf ? 8 : 0;
            #pragma unroll
            for (int r = 0; r < 8; ++r) {
                const float hv = fmaxf(a.f[r] + b1v[j], 0.0f);
                myLds[(mbase + r) * 64 + col] = (_Float16)hv;   // D -> row-major LDS
            }
        }

        // Write phase (h1) -> read phase (a20/a21).
        LDS_PHASE_FENCE();

        // --- layer 2: h2 = relu(h1 @ W2 + b2); K=64 -> 2 K-steps ----------------
        const HFrag a20 = loadA(0);
        const HFrag a21 = loadA(32);
        FAcc h2acc[4];
        #pragma unroll
        for (int j = 0; j < 4; ++j) {
            v8f c = {};
            c = __builtin_amdgcn_wmma_f32_16x16x32_f16(false, a20.v, false, w2f[0][j].v,
                                                       (short)0, c, false, false);
            c = __builtin_amdgcn_wmma_f32_16x16x32_f16(false, a21.v, false, w2f[1][j].v,
                                                       (short)0, c, false, false);
            h2acc[j].v = c;
        }
        #pragma unroll
        for (int j = 0; j < 4; ++j) {
            const int col   = 16 * j + nlo;
            const int mbase = hiHalf ? 8 : 0;
            #pragma unroll
            for (int r = 0; r < 8; ++r) {
                const float hv = fmaxf(h2acc[j].f[r] + b2v[j], 0.0f);
                myLds[(mbase + r) * 64 + col] = (_Float16)hv;
            }
        }

        // Write phase (h2) -> read phase (a30/a31).
        LDS_PHASE_FENCE();

        // --- layer 3: vel = (h2 @ W3 + b3) / viscosity; N padded 3 -> 16 --------
        const HFrag a30 = loadA(0);
        const HFrag a31 = loadA(32);
        v8f c3 = {};
        c3 = __builtin_amdgcn_wmma_f32_16x16x32_f16(false, a30.v, false, w3f[0].v,
                                                    (short)0, c3, false, false);
        c3 = __builtin_amdgcn_wmma_f32_16x16x32_f16(false, a31.v, false, w3f[1].v,
                                                    (short)0, c3, false, false);
        FAcc vel; vel.v = c3;

        // --- scatter-add to target nodes (lanes holding columns 0..2 only) ------
        if (nlo < 3) {
            const int mbase = hiHalf ? 8 : 0;
            #pragma unroll
            for (int r = 0; r < 8; ++r) {
                const int e = e0 + mbase + r;
                if (e < N_EDGES_C) {
                    const float v = (vel.f[r] + b3v) * invVisc;
                    atomicAdd(&out[3 * tgt[e] + nlo], v);
                }
            }
        }
    }
}

extern "C" void kernel_launch(void* const* d_in, const int* in_sizes, int n_in,
                              void* d_out, int out_size, void* d_ws, size_t ws_size,
                              hipStream_t stream) {
    const float* rel   = (const float*)d_in[0];
    const int*   tgt   = (const int*)d_in[1];
    const int*   src   = (const int*)d_in[2];
    const float* force = (const float*)d_in[3];
    const float* visc  = (const float*)d_in[4];
    const float* med   = (const float*)d_in[5];
    const float* cd    = (const float*)d_in[6];
    const float* W1    = (const float*)d_in[7];
    const float* b1    = (const float*)d_in[8];
    const float* W2    = (const float*)d_in[9];
    const float* b2    = (const float*)d_in[10];
    const float* W3    = (const float*)d_in[11];
    const float* b3    = (const float*)d_in[12];
    float* out = (float*)d_out;

    (void)d_ws; (void)ws_size; (void)n_in; (void)in_sizes;

    // d_out is poison-filled by the harness -> zero before scatter-adds.
    zero_out_kernel<<<(out_size + 255) / 256, 256, 0, stream>>>(out, out_size);

    // 1024 blocks x 8 waves = 8192 persistent waves; ~24 tiles each of 200000.
    pairvel_wmma_kernel<<<1024, 256, 0, stream>>>(rel, tgt, src, force, visc, med, cd,
                                                  W1, b1, W2, b2, W3, b3, out);
}
